// Context_Interaction_Block_14405320311344
// MI455X (gfx1250) — compile-verified, matrix operand order checked by reading, not run on hardware
//
#include <hip/hip_runtime.h>
#include <hip/hip_bf16.h>
#include <math.h>

// ---------------------------------------------------------------------------
// Swin-style window-attention block for MI455X (gfx1250).
// - all matmuls on v_wmma_f32_16x16x32_bf16 (wave32)
// - GEMM weight tiles staged in LDS via Tensor Data Mover (tensor_load_to_lds,
//   TENSORcnt, double-buffered) when the builtin is available
// ---------------------------------------------------------------------------

typedef __bf16 bf16_t;
typedef __attribute__((ext_vector_type(16))) __bf16 v16bf;
typedef __attribute__((ext_vector_type(8)))  __bf16 v8bf;
typedef __attribute__((ext_vector_type(8)))  float  v8f;
typedef unsigned int u32x4 __attribute__((ext_vector_type(4)));
typedef int          i32x4 __attribute__((ext_vector_type(4)));
typedef int          i32x8 __attribute__((ext_vector_type(8)));

#define DIM      256
#define HEADS    8
#define NTOK     64          // tokens per window
#define HEAD_DIM 32
#define HIDDEN   1024
#define TOKENS   131072      // 32*64*64
#define NWIN     2048
#define SCALE    0.17677669529663687f   // 32^-0.5
#define EPS      1e-5f

#if defined(__has_builtin)
#if __has_builtin(__builtin_amdgcn_tensor_load_to_lds) && \
    __has_builtin(__builtin_amdgcn_s_wait_tensorcnt)
#define HAVE_TDM 1
#endif
#endif

// windowed token index -> original row index in (B,H,W) token order
__device__ __forceinline__ size_t orig_index(int t) {
    int n   = t & 63;
    int win = t >> 6;
    int ww  = win & 7;
    int wh  = (win >> 3) & 7;
    int b   = win >> 6;
    int i   = n >> 3, j = n & 7;
    return ((size_t)b * 64 + (wh * 8 + i)) * 64 + (ww * 8 + j);
}

// 16x32 bf16 fragment (A-layout) from row-major [rows x ld] (global or LDS).
// lanes 0-15 -> M=lane, K={0..7,16..23}; lanes 16-31 -> M=lane-16, K={8..15,24..31}
__device__ __forceinline__ v16bf load_frag(const bf16_t* base, int ld, int r0, int k0) {
    const int lane  = threadIdx.x & 31;
    const int m     = lane & 15;
    const int khalf = (lane & 16) ? 8 : 0;
    const bf16_t* p = base + (size_t)(r0 + m) * ld + k0 + khalf;
    v8bf lo = *(const v8bf*)p;
    v8bf hi = *(const v8bf*)(p + 16);
    v16bf r;
#pragma unroll
    for (int i = 0; i < 8; ++i) { r[i] = lo[i]; r[i + 8] = hi[i]; }
    return r;
}

__device__ __forceinline__ float gelu_exact(float x) {
    return 0.5f * x * (1.0f + erff(x * 0.70710678118654752f));
}

#ifdef HAVE_TDM
// Issue a TDM load of a 64(rows) x 64(k) bf16 tile of Bt[nrows x ktot] into LDS.
// D# layout per CDNA5 ISA (group0 + group1; groups 2/3 zero).
__device__ __forceinline__ void tdm_issue(const bf16_t* gsrc, unsigned lds_off,
                                          int ktot, int nrows) {
    unsigned long long ga = (unsigned long long)(size_t)gsrc;
    u32x4 g0;
    g0[0] = 1u;                                     // descriptor count = 1
    g0[1] = lds_off;                                // lds_addr
    g0[2] = (unsigned)(ga & 0xffffffffull);         // global_addr[31:0]
    g0[3] = (unsigned)((ga >> 32) & 0x01ffffffull)  // global_addr[56:32]
            | 0x80000000u;                          // type = 2 ("image")
    i32x8 g1;
    g1[0] = 1 << 16;                                          // data_size = 2B
    g1[1] = (int)(((unsigned)ktot & 0xffffu) << 16);          // tensor_dim0 lo16
    g1[2] = (int)(((unsigned)ktot >> 16)
                  | (((unsigned)nrows & 0xffffu) << 16));     // dim0 hi | dim1 lo
    g1[3] = (int)(((unsigned)nrows >> 16) | (64u << 16));     // dim1 hi | tile_dim0=64
    g1[4] = 64;                                               // tile_dim1=64
    g1[5] = ktot;                                             // dim0_stride lo32
    g1[6] = 0;
    g1[7] = 0;
    i32x4 gz = {0, 0, 0, 0};
#if __clang_major__ >= 23
    i32x8 gz8 = {0, 0, 0, 0, 0, 0, 0, 0};
    __builtin_amdgcn_tensor_load_to_lds(g0, g1, gz, gz, gz8, 0);
#else
    __builtin_amdgcn_tensor_load_to_lds(g0, g1, gz, gz, 0);
#endif
}
#endif

// ---------------------------------------------------------------------------
// Weight transpose + fp32 -> bf16 convert: wt[n*K + k] = w[k*N + n]
// ---------------------------------------------------------------------------
__global__ void wcvt_kernel(const float* __restrict__ w, bf16_t* __restrict__ wt,
                            int K, int N) {
    int idx = blockIdx.x * 256 + threadIdx.x;
    if (idx >= K * N) return;
    int k = idx / N, n = idx % N;
    wt[(size_t)n * K + k] = (bf16_t)w[idx];
}

// ---------------------------------------------------------------------------
// LayerNorm over C=256, one wave per token, optional window-partition gather.
// ---------------------------------------------------------------------------
template <bool PERMUTE>
__global__ void layernorm_kernel(const float* __restrict__ x,
                                 const float* __restrict__ g,
                                 const float* __restrict__ b,
                                 bf16_t* __restrict__ out) {
    int token = blockIdx.x * 8 + (threadIdx.x >> 5);
    int lane  = threadIdx.x & 31;
    size_t src = PERMUTE ? orig_index(token) : (size_t)token;
    const float* p = x + src * DIM + lane * 8;

    float v[8];
    float s = 0.f, sq = 0.f;
#pragma unroll
    for (int i = 0; i < 8; ++i) { v[i] = p[i]; s += v[i]; sq += v[i] * v[i]; }
#pragma unroll
    for (int off = 16; off >= 1; off >>= 1) {
        s  += __shfl_xor(s,  off, 32);
        sq += __shfl_xor(sq, off, 32);
    }
    float mean = s * (1.f / DIM);
    float rstd = rsqrtf(sq * (1.f / DIM) - mean * mean + EPS);

    bf16_t* o = out + (size_t)token * DIM + lane * 8;
#pragma unroll
    for (int i = 0; i < 8; ++i) {
        int c = lane * 8 + i;
        o[i] = (bf16_t)((v[i] - mean) * rstd * g[c] + b[c]);
    }
}

// ---------------------------------------------------------------------------
// WMMA GEMM: C[M,N] = A[M,K] @ Bt[N,K]^T + bias.
// Block = 256 threads (8 waves); macro-tile 128 rows x 64 cols; each wave owns
// a 16x64 strip (A fragment reused across 4 WMMAs). Bt tiles (64x64, k-chunked)
// are staged in LDS: TDM double-buffered when available, else cooperative loads.
// Epilogues: 0 = bf16 (QKV); 1 = GELU->bf16 (FC1);
//            2 = f32 + shortcut gather (proj); 3 = f32 + residual scatter (FC2)
// ---------------------------------------------------------------------------
template <int EPI>
__global__ void gemm_wmma_kernel(const bf16_t* __restrict__ A,
                                 const bf16_t* __restrict__ Bt,
                                 const float* __restrict__ bias,
                                 bf16_t* __restrict__ outb,
                                 float* __restrict__ outf,
                                 const float* __restrict__ resid,
                                 int M, int N, int K) {
    __shared__ bf16_t btile[2][64][64];   // 2 x 8 KB

    const int wid  = threadIdx.x >> 5;
    const int lane = threadIdx.x & 31;
    const int row0 = blockIdx.x * 128 + wid * 16;
    const int col0 = blockIdx.y * 64;
    const int NC   = K / 64;

    v8f acc[4] = {{}, {}, {}, {}};

#ifdef HAVE_TDM
    if (wid == 0)
        tdm_issue(Bt + (size_t)col0 * K, (unsigned)(size_t)&btile[0][0][0], K, N - col0);
    for (int c = 0; c < NC; ++c) {
        if (wid == 0) {
            if (c + 1 < NC) {
                tdm_issue(Bt + (size_t)col0 * K + (c + 1) * 64,
                          (unsigned)(size_t)&btile[(c + 1) & 1][0][0], K, N - col0);
                __builtin_amdgcn_s_wait_tensorcnt(1);   // chunk c landed
            } else {
                __builtin_amdgcn_s_wait_tensorcnt(0);
            }
        }
        __syncthreads();
        const bf16_t* bb = &btile[c & 1][0][0];
#else
    for (int c = 0; c < NC; ++c) {
        __syncthreads();   // protect previous chunk's readers
        {   // cooperative load: 64 rows x 64 k, 16 elems per thread
            int r  = threadIdx.x >> 2;
            int cb = (threadIdx.x & 3) * 16;
            const bf16_t* src = Bt + (size_t)(col0 + r) * K + c * 64 + cb;
            *(v8bf*)&btile[0][r][cb]     = *(const v8bf*)src;
            *(v8bf*)&btile[0][r][cb + 8] = *(const v8bf*)(src + 8);
        }
        __syncthreads();
        const bf16_t* bb = &btile[0][0][0];
#endif
#pragma unroll
        for (int kk = 0; kk < 64; kk += 32) {
            if (c * 64 + kk + 32 < K)
                __builtin_prefetch(A + (size_t)(row0 + (lane & 15)) * K + c * 64 + kk + 32, 0, 0);
            v16bf a = load_frag(A, K, row0, c * 64 + kk);
#pragma unroll
            for (int t = 0; t < 4; ++t) {
                v16bf b = load_frag(bb, 64, t * 16, kk);
                acc[t] = __builtin_amdgcn_wmma_f32_16x16x32_bf16(
                    false, a, false, b, (short)0, acc[t], false, false);
            }
        }
#ifdef HAVE_TDM
        __syncthreads();   // all readers done before wave0 re-targets this buffer
#endif
    }

    const int mbase = (lane & 16) ? 8 : 0;
    const int n     = lane & 15;
#pragma unroll
    for (int t = 0; t < 4; ++t) {
        int col  = col0 + t * 16 + n;
        float bv = bias[col];
#pragma unroll
        for (int r = 0; r < 8; ++r) {
            int row = row0 + mbase + r;
            float v = acc[t][r] + bv;
            if (EPI == 0) {
                outb[(size_t)row * N + col] = (bf16_t)v;
            } else if (EPI == 1) {
                outb[(size_t)row * N + col] = (bf16_t)gelu_exact(v);
            } else if (EPI == 2) {
                size_t g = orig_index(row);
                outf[(size_t)row * DIM + col] = resid[g * DIM + col] + v;
            } else {
                size_t g = orig_index(row);
                outf[g * DIM + col] = resid[(size_t)row * DIM + col] + v;
            }
        }
    }
}

// ---------------------------------------------------------------------------
// Window attention: 1 block per (window, head); 4 waves, one 16-row strip each.
//   S = (q*SCALE) @ k^T + rel_bias ; P = softmax(S) ; O = P @ v
// qkv row layout per token (768): [q | k | v][head][d].
// V is staged once per block into LDS as V^T so P@V B-fragments are LDS reads.
// ---------------------------------------------------------------------------
__global__ void attn_kernel(const bf16_t* __restrict__ qkv,
                            const float* __restrict__ bias_table,
                            bf16_t* __restrict__ out) {
    __shared__ bf16_t plds[4][16][64];   // per-wave P tile (C-layout -> A-layout)
    __shared__ bf16_t vT[HEAD_DIM][NTOK];

    const int wid  = threadIdx.x >> 5;
    const int lane = threadIdx.x & 31;
    const int win  = blockIdx.x >> 3;
    const int head = blockIdx.x & 7;

    const bf16_t* base = qkv + (size_t)win * NTOK * 768 + head * HEAD_DIM;
    const bf16_t* qb = base;
    const bf16_t* kb = base + DIM;
    const bf16_t* vb = base + 2 * DIM;

    // stage V^T: each of 128 threads moves 16 contiguous bf16 of one v row
    {
        int tok   = threadIdx.x >> 1;
        int dbase = (threadIdx.x & 1) * 16;
        v8bf lo = *(const v8bf*)(vb + (size_t)tok * 768 + dbase);
        v8bf hi = *(const v8bf*)(vb + (size_t)tok * 768 + dbase + 8);
#pragma unroll
        for (int i = 0; i < 8; ++i) {
            vT[dbase + i][tok]     = lo[i];
            vT[dbase + 8 + i][tok] = hi[i];
        }
    }

    // ---- S = q @ k^T (K = head_dim = 32: one WMMA per 16x16 tile) ----
    v16bf aq = load_frag(qb, 768, wid * 16, 0);
    v8f s[4];
#pragma unroll
    for (int ct = 0; ct < 4; ++ct) {
        v16bf bk = load_frag(kb, 768, ct * 16, 0);
        v8f z = {};
        s[ct] = __builtin_amdgcn_wmma_f32_16x16x32_bf16(
            false, aq, false, bk, (short)0, z, false, false);
    }

    // ---- bias + scale + softmax across 64 keys per query row ----
    const int mbase = (lane & 16) ? 8 : 0;
    const int cl    = lane & 15;
    float p[4][8];
#pragma unroll
    for (int e = 0; e < 8; ++e) {
        int nrow = wid * 16 + mbase + e;
        int i = nrow >> 3, j = nrow & 7;
        float mx = -3.0e38f;
#pragma unroll
        for (int ct = 0; ct < 4; ++ct) {
            int m  = ct * 16 + cl;
            int i2 = m >> 3, j2 = m & 7;
            int ridx = (i - i2 + 7) * 15 + (j - j2 + 7);
            float lg = s[ct][e] * SCALE + bias_table[ridx * HEADS + head];
            p[ct][e] = lg;
            mx = fmaxf(mx, lg);
        }
#pragma unroll
        for (int off = 8; off >= 1; off >>= 1)    // row lives in one 16-lane half
            mx = fmaxf(mx, __shfl_xor(mx, off, 32));
        float sum = 0.f;
#pragma unroll
        for (int ct = 0; ct < 4; ++ct) { p[ct][e] = __expf(p[ct][e] - mx); sum += p[ct][e]; }
#pragma unroll
        for (int off = 8; off >= 1; off >>= 1)
            sum += __shfl_xor(sum, off, 32);
        float rs = 1.f / sum;
#pragma unroll
        for (int ct = 0; ct < 4; ++ct)
            plds[wid][mbase + e][ct * 16 + cl] = (bf16_t)(p[ct][e] * rs);
    }
    __syncthreads();   // covers both plds and vT

    // ---- O = P @ v (K = 64: two WMMA accumulations per 16x16 tile) ----
#pragma unroll
    for (int nt = 0; nt < 2; ++nt) {
        v8f o = {};
#pragma unroll
        for (int kt = 0; kt < 2; ++kt) {
            v16bf ap = load_frag(&plds[wid][0][0], 64, 0, kt * 32);
            v16bf bv = load_frag(&vT[0][0],        64, nt * 16, kt * 32);
            o = __builtin_amdgcn_wmma_f32_16x16x32_bf16(
                false, ap, false, bv, (short)0, o, false, false);
        }
#pragma unroll
        for (int e = 0; e < 8; ++e) {
            int token = win * NTOK + wid * 16 + mbase + e;
            int col   = head * HEAD_DIM + nt * 16 + cl;
            out[(size_t)token * DIM + col] = (bf16_t)o[e];
        }
    }
}

// ---------------------------------------------------------------------------
// Host-side orchestration
// ---------------------------------------------------------------------------
extern "C" void kernel_launch(void* const* d_in, const int* in_sizes, int n_in,
                              void* d_out, int out_size, void* d_ws, size_t ws_size,
                              hipStream_t stream) {
    const float* x        = (const float*)d_in[0];
    const float* norm1_g  = (const float*)d_in[2];
    const float* norm1_b  = (const float*)d_in[3];
    const float* qkv_w    = (const float*)d_in[4];
    const float* qkv_b    = (const float*)d_in[5];
    const float* rel_tab  = (const float*)d_in[6];
    const float* proj_w   = (const float*)d_in[7];
    const float* proj_b   = (const float*)d_in[8];
    const float* norm2_g  = (const float*)d_in[9];
    const float* norm2_b  = (const float*)d_in[10];
    const float* fc1_w    = (const float*)d_in[11];
    const float* fc1_b    = (const float*)d_in[12];
    const float* fc2_w    = (const float*)d_in[13];
    const float* fc2_b    = (const float*)d_in[14];
    float* out = (float*)d_out;

    char* ws = (char*)d_ws;
    const size_t T = TOKENS;
    size_t off = 0;
    bf16_t* xw   = (bf16_t*)(ws + off); off += T * DIM * 2;          //  64 MB (reused as xn2)
    bf16_t* qkvb = (bf16_t*)(ws + off); off += T * 3 * DIM * 2;      // 192 MB
    bf16_t* attn = (bf16_t*)(ws + off); off += T * DIM * 2;          //  64 MB
    float*  x1   = (float*)(ws + off);  off += T * DIM * 4;          // 128 MB
    bf16_t* qkv_wT  = (bf16_t*)(ws + off); off += (size_t)768 * DIM * 2;
    bf16_t* proj_wT = (bf16_t*)(ws + off); off += (size_t)DIM * DIM * 2;
    bf16_t* fc1_wT  = (bf16_t*)(ws + off); off += (size_t)HIDDEN * DIM * 2;
    bf16_t* fc2_wT  = (bf16_t*)(ws + off); off += (size_t)DIM * HIDDEN * 2;
    bf16_t* hbuf = qkvb;   // 256 MB region spanning qkvb+attn, free after proj

    // 1) weights -> bf16, transposed to N-major
    wcvt_kernel<<<(DIM * 768 + 255) / 256, 256, 0, stream>>>(qkv_w, qkv_wT, DIM, 768);
    wcvt_kernel<<<(DIM * DIM + 255) / 256, 256, 0, stream>>>(proj_w, proj_wT, DIM, DIM);
    wcvt_kernel<<<(DIM * HIDDEN + 255) / 256, 256, 0, stream>>>(fc1_w, fc1_wT, DIM, HIDDEN);
    wcvt_kernel<<<(HIDDEN * DIM + 255) / 256, 256, 0, stream>>>(fc2_w, fc2_wT, HIDDEN, DIM);

    // 2) LN1 fused with window partition
    layernorm_kernel<true><<<T / 8, 256, 0, stream>>>(x, norm1_g, norm1_b, xw);

    // 3) QKV GEMM: (T x 256) @ (256 x 768)
    gemm_wmma_kernel<0><<<dim3(T / 128, 768 / 64), 256, 0, stream>>>(
        xw, qkv_wT, qkv_b, qkvb, nullptr, nullptr, (int)T, 768, DIM);

    // 4) window attention
    attn_kernel<<<NWIN * HEADS, 128, 0, stream>>>(qkvb, rel_tab, attn);

    // 5) proj GEMM + shortcut (gather from orig-order x) -> x1 (window order)
    gemm_wmma_kernel<2><<<dim3(T / 128, DIM / 64), 256, 0, stream>>>(
        attn, proj_wT, proj_b, nullptr, x1, x, (int)T, DIM, DIM);

    // 6) LN2: x1 -> xw
    layernorm_kernel<false><<<T / 8, 256, 0, stream>>>(x1, norm2_g, norm2_b, xw);

    // 7) FC1 + GELU
    gemm_wmma_kernel<1><<<dim3(T / 128, HIDDEN / 64), 256, 0, stream>>>(
        xw, fc1_wT, fc1_b, hbuf, nullptr, nullptr, (int)T, HIDDEN, DIM);

    // 8) FC2 + residual, window-reverse scatter into d_out
    gemm_wmma_kernel<3><<<dim3(T / 128, DIM / 64), 256, 0, stream>>>(
        hbuf, fc2_wT, fc2_b, nullptr, out, x1, (int)T, DIM, HIDDEN);

    (void)in_sizes; (void)n_in; (void)out_size; (void)ws_size;
}
